// UpdateModule_42245298323616
// MI455X (gfx1250) — matrix operand order repaired and checked by reading, not compiled, not run on hardware
//
#include <hip/hip_runtime.h>
#include <math.h>

typedef __attribute__((ext_vector_type(16))) _Float16 v16h;
typedef __attribute__((ext_vector_type(8)))  float    v8f;

#define BDIM 64
#define SDIM 48
#define NDIM 5
#define SLOT 256
#define ENT  128
#define ROLE 64
#define MS   (BDIM * SDIM)          // 3072 rows for the activation GEMMs
#define EPSV 1e-6f

// ---------------- workspace layout (bytes) ----------------
// projh : [5][MS][SLOT] fp16   @ 0              = 7,864,320 B
// Wh    : 131072 fp16 (We1,We2,Wr1,Wr2,Wr3 pad) @ 7,864,320 = 262,144 B
// acts  : e1|e2|r1|r2|r3 fp32                   @ 8,126,464 = 5,505,024 B
#define PROJH_ELEMS   (5 * MS * SLOT)         // 3,932,160
#define WH_ELEMS      131072
#define WE1_OFF 0
#define WE2_OFF 32768
#define WR1_OFF 65536
#define WR2_OFF 81920
#define WR3_OFF 98304
// acts offsets (floats)
#define E1_OFF 0
#define E2_OFF (MS * ENT)                     // 393216
#define R1_OFF (2 * MS * ENT)                 // 786432
#define R2_OFF (2 * MS * ENT + MS * ROLE)     // 983040
#define R3_OFF (2 * MS * ENT + 2 * MS * ROLE) // 1179648

// ---------------- Kernel A: projections -> fp16 ----------------
// proj_k = eps * sum_n slots[:,n,:] + (1-5eps) * slots[:,k,:]
__global__ void proj_kernel(const float* __restrict__ slots,
                            _Float16* __restrict__ projh) {
    int idx = blockIdx.x * blockDim.x + threadIdx.x;  // over MS*SLOT
    if (idx >= MS * SLOT) return;
    int m = idx >> 8;          // row (b*48+s)
    int c = idx & 255;         // slot channel
    const float* p = slots + (size_t)m * NDIM * SLOT + c;
    float s0 = p[0 * SLOT], s1 = p[1 * SLOT], s2 = p[2 * SLOT],
          s3 = p[3 * SLOT], s4 = p[4 * SLOT];
    float base = EPSV * (s0 + s1 + s2 + s3 + s4);
    float w = 1.0f - 5.0f * EPSV;
    projh[(size_t)0 * MS * SLOT + idx] = (_Float16)(base + w * s0);
    projh[(size_t)1 * MS * SLOT + idx] = (_Float16)(base + w * s1);
    projh[(size_t)2 * MS * SLOT + idx] = (_Float16)(base + w * s2);
    projh[(size_t)3 * MS * SLOT + idx] = (_Float16)(base + w * s3);
    projh[(size_t)4 * MS * SLOT + idx] = (_Float16)(base + w * s4);
}

// ---------------- Kernel B: f32 -> f16 weight convert ----------------
__global__ void cvt_kernel(const float* __restrict__ src,
                           _Float16* __restrict__ dst, int n) {
    int i = blockIdx.x * blockDim.x + threadIdx.x;
    if (i < n) dst[i] = (_Float16)src[i];
}

// ---------------- Kernel C: activation GEMMs via WMMA f16 ----------------
// One wave per 16x16 output tile. out = tanh(proj @ W^T + b).
// tiles per m-tile: e1(8) e2(8) r1(4) r2(4) r3(4) = 28; m-tiles = 192; total 5376 waves.
__global__ void act_gemm_kernel(const _Float16* __restrict__ projh,
                                const _Float16* __restrict__ Wh,
                                const float* __restrict__ b_e1,
                                const float* __restrict__ b_e2,
                                const float* __restrict__ b_r1,
                                const float* __restrict__ b_r2,
                                const float* __restrict__ b_r3,
                                float* __restrict__ acts) {
    int gtid = blockIdx.x * blockDim.x + threadIdx.x;
    int wave = gtid >> 5;
    int lane = threadIdx.x & 31;
    if (wave >= 192 * 28) return;

    int mt  = wave / 28;
    int rem = wave % 28;

    int act, nt, Nw;
    const _Float16* Wsub;
    const float* bias;
    float* outp;
    if (rem < 8)       { act = 0; nt = rem;      Nw = ENT;  Wsub = Wh + WE1_OFF; bias = b_e1; outp = acts + E1_OFF; }
    else if (rem < 16) { act = 1; nt = rem - 8;  Nw = ENT;  Wsub = Wh + WE2_OFF; bias = b_e2; outp = acts + E2_OFF; }
    else if (rem < 20) { act = 2; nt = rem - 16; Nw = ROLE; Wsub = Wh + WR1_OFF; bias = b_r1; outp = acts + R1_OFF; }
    else if (rem < 24) { act = 3; nt = rem - 20; Nw = ROLE; Wsub = Wh + WR2_OFF; bias = b_r2; outp = acts + R2_OFF; }
    else               { act = 4; nt = rem - 24; Nw = ROLE; Wsub = Wh + WR3_OFF; bias = b_r3; outp = acts + R3_OFF; }

    const _Float16* P = projh + (size_t)act * MS * SLOT;
    int mrow  = mt * 16 + (lane & 15);   // A-matrix row (M) for this lane
    int ncol  = nt * 16 + (lane & 15);   // B/C/D column (N) for this lane
    int koffa = (lane < 16) ? 0 : 8;     // ISA 16-bit A layout lane-half K offset
    int koffb = (lane < 16) ? 0 : 16;    // unused: B is loaded with koffa, same K order as A

    v8f c = {};
    for (int k0 = 0; k0 < SLOT; k0 += 32) {
        v16h a, b;
#pragma unroll
        for (int jj = 0; jj < 8; ++jj) {
            // A (16x32, MxK): VGPR jj<4 -> K = 2jj + {0|8}; jj>=4 -> K = 16 + 2(jj-4) + {0|8}
            int ka = k0 + ((jj < 4) ? (2 * jj + koffa) : (16 + 2 * (jj - 4) + koffa));
            a[2 * jj]     = P[(size_t)mrow * SLOT + ka];
            a[2 * jj + 1] = P[(size_t)mrow * SLOT + ka + 1];
            // B (32x16, KxN): same K order as A; B[k][n] = W[n][k]
            int kb = k0 + ((jj < 4) ? (2 * jj + koffa) : (16 + 2 * (jj - 4) + koffa));
            b[2 * jj]     = Wsub[(size_t)ncol * SLOT + kb];
            b[2 * jj + 1] = Wsub[(size_t)ncol * SLOT + kb + 1];
        }
        c = __builtin_amdgcn_wmma_f32_16x16x32_f16(false, a, false, b,
                                                   (short)0, c, false, false);
    }

    float bn = bias[ncol];
#pragma unroll
    for (int i = 0; i < 8; ++i) {
        int row = mt * 16 + i + ((lane < 16) ? 0 : 8);  // C/D: M = i (+8 for upper lanes)
        outp[(size_t)row * Nw + ncol] = tanhf(c[i] + bn);
    }
}

// ---------------- Kernel D: the 48-step scan, TPR resident in 320KB LDS ----
// Partition TPR along the final ENT axis (f) in blocks of 8: every contraction
// and update is then local to the workgroup. Chunk = 128*64*8 f32 = 256 KB LDS.
// grid = 64 batches * 16 f-blocks; block = 512 threads (one per (r,f) pair).
#define FB 8
#define NF (ENT / FB)            // 16
#define CHUNK (ENT * ROLE * FB)  // 65536 floats
// LDS float offsets
#define L_U    (CHUNK)                // u[64][8]
#define L_V    (L_U + 512)
#define L_CA   (L_V + 512)
#define L_CB   (L_CA + 512)
#define L_E1   (L_CB + 512)           // 128
#define L_E2   (L_E1 + 128)           // 128
#define L_R1   (L_E2 + 128)           // 64
#define L_R2   (L_R1 + 64)
#define L_R3   (L_R2 + 64)
#define L_HAT  (L_R3 + 64)            // 24: w[0..7] m[8..15] b[16..23]
#define L_TOT  (L_HAT + 24)           // 68056 floats = 272,224 B  (< 320 KB)

__global__ void scan_kernel(const float* __restrict__ acts,
                            float* __restrict__ out) {
    extern __shared__ float sm[];
    float* TPR = sm;

    const int b     = blockIdx.x >> 4;
    const int fb    = blockIdx.x & (NF - 1);
    const int fbase = fb * FB;
    const int tid   = threadIdx.x;      // 0..511
    const int r     = tid >> 3;
    const int f     = tid & 7;

    for (int i = tid; i < L_TOT; i += 512) sm[i] = 0.0f;
    __syncthreads();

    for (int s = 0; s < SDIM; ++s) {
        const int m = b * SDIM + s;
        // stage this step's activation vectors into LDS
        if (tid < 128)      sm[L_E1 + tid]         = acts[E1_OFF + (size_t)m * ENT + tid];
        else if (tid < 256) sm[L_E2 + (tid - 128)] = acts[E2_OFF + (size_t)m * ENT + (tid - 128)];
        else if (tid < 320) sm[L_R1 + (tid - 256)] = acts[R1_OFF + (size_t)m * ROLE + (tid - 256)];
        else if (tid < 384) sm[L_R2 + (tid - 320)] = acts[R2_OFF + (size_t)m * ROLE + (tid - 320)];
        else if (tid < 448) sm[L_R3 + (tid - 384)] = acts[R3_OFF + (size_t)m * ROLE + (tid - 384)];
        __syncthreads();

        // u[r,f] = sum_e e1[e]*TPR[e,r,f]; v[r,f] = sum_e e2[e]*TPR[e,r,f]
        float au = 0.0f, av = 0.0f;
#pragma unroll 4
        for (int e = 0; e < ENT; ++e) {
            float t = TPR[e * 512 + tid];          // stride-1 across lanes: conflict-free
            au = fmaf(sm[L_E1 + e], t, au);        // broadcast reads
            av = fmaf(sm[L_E2 + e], t, av);
        }
        sm[L_U + tid] = au;
        sm[L_V + tid] = av;
        __syncthreads();

        // hats: contract u/v over r (tiny; 8 lanes)
        if (tid < 8) {
            float wh = 0.0f, mh = 0.0f, bh = 0.0f;
#pragma unroll 4
            for (int rr = 0; rr < ROLE; ++rr) {
                float uu = sm[L_U + rr * 8 + tid];
                float vv = sm[L_V + rr * 8 + tid];
                wh = fmaf(sm[L_R1 + rr], uu, wh);
                mh = fmaf(sm[L_R2 + rr], uu, mh);
                bh = fmaf(sm[L_R3 + rr], vv, bh);
            }
            sm[L_HAT + tid]      = wh;
            sm[L_HAT + 8 + tid]  = mh;
            sm[L_HAT + 16 + tid] = bh;
        }
        __syncthreads();

        // cA[r,f] = r1*(e2[f]-w) + r2*(w-m);  cB[r,f] = r3*(e1[f]-b)
        {
            float wh = sm[L_HAT + f], mh = sm[L_HAT + 8 + f], bh = sm[L_HAT + 16 + f];
            float R1v = sm[L_R1 + r], R2v = sm[L_R2 + r], R3v = sm[L_R3 + r];
            sm[L_CA + tid] = R1v * (sm[L_E2 + fbase + f] - wh) + R2v * (wh - mh);
            sm[L_CB + tid] = R3v * (sm[L_E1 + fbase + f] - bh);
        }
        __syncthreads();

        // TPR[e,r,f] += clip(e1[e]*cA + e2[e]*cB, -1, 1)
        float ca = sm[L_CA + tid];
        float cb = sm[L_CB + tid];
#pragma unroll 4
        for (int e = 0; e < ENT; ++e) {
            float d = fmaf(sm[L_E1 + e], ca, sm[L_E2 + e] * cb);
            d = fminf(1.0f, fmaxf(-1.0f, d));
            TPR[e * 512 + tid] += d;
        }
        __syncthreads();
    }

    // write final TPR chunk: out[b][e][r][fbase+f]
    for (int e = 0; e < ENT; ++e) {
        size_t oi = ((((size_t)b * ENT + e) * ROLE + r) * ENT) + fbase + f;
        out[oi] = TPR[e * 512 + tid];
    }
}

// ---------------- host-side launch ----------------
extern "C" void kernel_launch(void* const* d_in, const int* in_sizes, int n_in,
                              void* d_out, int out_size, void* d_ws, size_t ws_size,
                              hipStream_t stream) {
    const float* slots = (const float*)d_in[0];
    const float* W_e1  = (const float*)d_in[1];
    const float* b_e1  = (const float*)d_in[2];
    const float* W_e2  = (const float*)d_in[3];
    const float* b_e2  = (const float*)d_in[4];
    const float* W_r1  = (const float*)d_in[5];
    const float* b_r1  = (const float*)d_in[6];
    const float* W_r2  = (const float*)d_in[7];
    const float* b_r2  = (const float*)d_in[8];
    const float* W_r3  = (const float*)d_in[9];
    const float* b_r3  = (const float*)d_in[10];
    float* out = (float*)d_out;

    char* wsb = (char*)d_ws;
    _Float16* projh = (_Float16*)wsb;
    _Float16* Wh    = projh + PROJH_ELEMS;
    float*    acts  = (float*)(wsb + (size_t)(PROJH_ELEMS + WH_ELEMS) * sizeof(_Float16));

    // A: projections -> fp16
    proj_kernel<<<(MS * SLOT + 255) / 256, 256, 0, stream>>>(slots, projh);
    // B: weights -> fp16
    cvt_kernel<<<(ENT * SLOT + 255) / 256, 256, 0, stream>>>(W_e1, Wh + WE1_OFF, ENT * SLOT);
    cvt_kernel<<<(ENT * SLOT + 255) / 256, 256, 0, stream>>>(W_e2, Wh + WE2_OFF, ENT * SLOT);
    cvt_kernel<<<(ROLE * SLOT + 255) / 256, 256, 0, stream>>>(W_r1, Wh + WR1_OFF, ROLE * SLOT);
    cvt_kernel<<<(ROLE * SLOT + 255) / 256, 256, 0, stream>>>(W_r2, Wh + WR2_OFF, ROLE * SLOT);
    cvt_kernel<<<(ROLE * SLOT + 255) / 256, 256, 0, stream>>>(W_r3, Wh + WR3_OFF, ROLE * SLOT);
    // C: WMMA activation GEMMs (5376 tiles, 8 waves/block)
    act_gemm_kernel<<<672, 256, 0, stream>>>(projh, Wh, b_e1, b_e2, b_r1, b_r2, b_r3, acts);
    // D: LDS-resident sequential scan (272,224 B dynamic LDS per WG; CDNA5 = 320 KB/WGP)
    scan_kernel<<<BDIM * NF, 512, L_TOT * sizeof(float), stream>>>(acts, out);
}
